// EchoStateReservoir_23295902613805
// MI455X (gfx1250) — matrix operand order repaired and checked by reading, not compiled, not run on hardware
//
#include <hip/hip_runtime.h>

typedef __bf16 bf16_t;
typedef __attribute__((ext_vector_type(8)))  __bf16 v8bf;
typedef __attribute__((ext_vector_type(16))) __bf16 v16bf;
typedef __attribute__((ext_vector_type(8)))  float  v8f;

#define BDIM   8
#define TSTEPS 4096
#define DMODEL 512
#define DRES   1024
#define MROWS  (BDIM * TSTEPS)   // 32768

// ---------------- WMMA fragment helpers (wave32, 16x16x32 bf16) ----------------
// A (16xK, M-major): lane l<16 -> row=l, K {kb..kb+15};
//                    lane l>=16 -> row=l-16, K {kb+16..kb+31}
__device__ __forceinline__ v16bf load_fragA(const bf16_t* __restrict__ base, int row,
                                            int ld, int kbase, int lane) {
  const bf16_t* p = base + (size_t)row * ld + kbase + ((lane >> 4) << 4);
  v8bf a = *(const v8bf*)(p);
  v8bf b = *(const v8bf*)(p + 8);
  v16bf r;
#pragma unroll
  for (int i = 0; i < 8; ++i) { r[i] = a[i]; r[i + 8] = b[i]; }
  return r;
}
// B stored as N x K row-major (i.e. W[n][k]): lane l -> col n=l&15,
// contiguous 16 K values starting at kbase + (l>=16 ? 16 : 0)
__device__ __forceinline__ v16bf load_fragB(const bf16_t* __restrict__ base, int nrow,
                                            int ld, int kbase, int lane) {
  return *(const v16bf*)(base + (size_t)nrow * ld + kbase + ((lane >> 4) << 4));
}
__device__ __forceinline__ v8f wmma_bf16(v16bf a, v16bf b, v8f c) {
  return __builtin_amdgcn_wmma_f32_16x16x32_bf16(false, a, false, b, (short)0, c, false, false);
}

// ---------------- spectral radius estimate (power iteration) ----------------
__global__ void __launch_bounds__(1024)
spectral_kernel(const float* __restrict__ W, const float* __restrict__ log_rho,
                const float* __restrict__ leak, float* __restrict__ out) {
  __shared__ float v[DRES];
  __shared__ float red[32];
  __shared__ float nrm_s;
  const int tid  = threadIdx.x;
  const int lane = tid & 31;
  const int wid  = tid >> 5;
  unsigned uh = (unsigned)tid * 2654435761u + 0x9e3779b9u;
  uh ^= uh >> 16; uh *= 2246822519u; uh ^= uh >> 13;
  v[tid] = ((float)(uh & 0xFFFFu) * (1.0f / 32768.0f)) - 1.0f;
  __syncthreads();
  const int NITER = 128, WIN = 64;
  float logacc = 0.0f;
  const float* __restrict__ row = W + (size_t)tid * DRES;
  for (int it = 0; it < NITER; ++it) {
    float s = 0.0f;
    for (int k = 0; k < DRES; k += 4) {
      s += row[k] * v[k];         s += row[k + 1] * v[k + 1];
      s += row[k + 2] * v[k + 2]; s += row[k + 3] * v[k + 3];
    }
    float ss = s * s;
#pragma unroll
    for (int off = 16; off > 0; off >>= 1) ss += __shfl_xor(ss, off, 32);
    if (lane == 0) red[wid] = ss;
    __syncthreads();
    if (tid == 0) {
      float tot = 0.0f;
#pragma unroll
      for (int i = 0; i < 32; ++i) tot += red[i];
      nrm_s = fmaxf(sqrtf(tot), 1e-30f);
    }
    __syncthreads();
    float nrm = nrm_s;
    if (tid == 0 && it >= NITER - WIN) logacc += logf(nrm);
    v[tid] = s / nrm;
    __syncthreads();
  }
  if (tid == 0) {
    float eig = expf(logacc / (float)WIN);
    float rho = 1.5f / (1.0f + expf(-log_rho[0]));
    out[0] = rho / fmaxf(eig, 1e-6f);                 // scale for W_res
    out[1] = 1.0f / (1.0f + expf(-leak[0]));          // lk
  }
}

// ---------------- fp32 -> split bf16 (hi + lo), optional device-side scale ----------------
__global__ void cvt_split_kernel(const float* __restrict__ src, bf16_t* __restrict__ hi,
                                 bf16_t* __restrict__ lo, int n,
                                 const float* __restrict__ scale_ptr) {
  int i = blockIdx.x * blockDim.x + threadIdx.x;
  int stride = gridDim.x * blockDim.x;
  float s = scale_ptr ? *scale_ptr : 1.0f;
  for (; i < n; i += stride) {
    float w = src[i] * s;
    bf16_t h = (bf16_t)w;
    hi[i] = h;
    lo[i] = (bf16_t)(w - (float)h);
  }
}

// ---------------- zero h double-buffers (incl. pad rows) + barrier state ----------------
__global__ void scan_init_kernel(bf16_t* __restrict__ h_hi, bf16_t* __restrict__ h_lo,
                                 unsigned* __restrict__ bar) {
  int i = blockIdx.x * blockDim.x + threadIdx.x;
  if (i < 2 * 16 * DRES) { h_hi[i] = (bf16_t)0.0f; h_lo[i] = (bf16_t)0.0f; }
  if (i == 0) { bar[0] = 0u; bar[1] = 0u; }
}

// ---------------- generic split-bf16 WMMA GEMM: C[M][N] = A[M][K] * B(NxK)^T ----------------
__global__ void __launch_bounds__(256)
gemm_bf16split_kernel(const bf16_t* __restrict__ Ahi, const bf16_t* __restrict__ Alo,
                      const bf16_t* __restrict__ Bhi, const bf16_t* __restrict__ Blo,
                      float* __restrict__ C, int M, int N, int K) {
  const int lane  = threadIdx.x & 31;
  const int gwave = (int)((blockIdx.x * blockDim.x + threadIdx.x) >> 5);
  const int ntiles = N >> 4;
  const int mt = gwave / ntiles;
  const int nt = gwave - mt * ntiles;
  if (mt >= (M >> 4)) return;
  const int arow = (mt << 4) + (lane & 15);
  const int nrow = (nt << 4) + (lane & 15);
  v8f c = {};
#pragma unroll 2
  for (int kb = 0; kb < K; kb += 32) {
    v16bf ah = load_fragA(Ahi, arow, K, kb, lane);
    v16bf al = load_fragA(Alo, arow, K, kb, lane);
    v16bf bh = load_fragB(Bhi, nrow, K, kb, lane);
    v16bf bl = load_fragB(Blo, nrow, K, kb, lane);
    c = wmma_bf16(ah, bh, c);   // hi*hi
    c = wmma_bf16(ah, bl, c);   // hi*lo
    c = wmma_bf16(al, bh, c);   // lo*hi
  }
  const int crow0 = (mt << 4) + ((lane >> 4) << 3);
  const int col   = (nt << 4) + (lane & 15);
#pragma unroll
  for (int v = 0; v < 8; ++v) C[(size_t)(crow0 + v) * N + col] = c[v];
}

// ---------------- persistent reservoir scan: 16 WGs (one cluster-width), 4096 steps ----------------
__global__ void __launch_bounds__(256, 1)
scan_kernel(const bf16_t* __restrict__ Whi, const bf16_t* __restrict__ Wlo,   // [1024][1024] N x K
            const float* __restrict__ u,                                       // [32768][1024]
            bf16_t* __restrict__ hbuf_hi, bf16_t* __restrict__ hbuf_lo,        // [2][16][1024]
            bf16_t* __restrict__ Hhi, bf16_t* __restrict__ Hlo,                // [32768][1024]
            const float* __restrict__ scalars, unsigned* __restrict__ bar) {
  const int tid   = threadIdx.x;
  const int wave  = tid >> 5;
  const int lane  = tid & 31;
  const int wg    = blockIdx.x;                  // 0..15
  const int ntile = (wg << 2) + (wave >> 1);     // global N-tile 0..63 (16 output channels each)
  const int khalf = wave & 1;                    // split K=1024 across wave pairs
  const float lk = scalars[1];
  const float om = 1.0f - lk;
  const int nrow = (ntile << 4) + (lane & 15);
  const int k0   = khalf * (DRES / 2);
  const bool fin = (khalf == 0) && (lane < 16);  // finalize lanes
  const int col  = (ntile << 4) + lane;          // finalize column (fin lanes only)
  const bool incluster = (__builtin_amdgcn_cluster_id_x() != 0);

  __shared__ float partial[8][8][32];            // [wave][cvgpr][lane], 8 KB

  // ---- hoist the wave's W_hi B-fragments into registers (loop-invariant, 128 VGPRs) ----
  v16bf whi[16];
#pragma unroll
  for (int i = 0; i < 16; ++i) whi[i] = load_fragB(Whi, nrow, DRES, k0 + 32 * i, lane);

  for (int t = 0; t < TSTEPS; ++t) {
    const bf16_t* hr_hi = hbuf_hi + (size_t)(t & 1) * (16 * DRES);
    const bf16_t* hr_lo = hbuf_lo + (size_t)(t & 1) * (16 * DRES);
    bf16_t* hw_hi = hbuf_hi + (size_t)((t + 1) & 1) * (16 * DRES);
    bf16_t* hw_lo = hbuf_lo + (size_t)((t + 1) & 1) * (16 * DRES);

    // ---- preload u_t and h_old early: independent of this step's WMMA ----
    float ureg[8], holdreg[8];
    if (fin) {
#pragma unroll
      for (int v = 0; v < 8; ++v) {
        const int m = v * TSTEPS + t;
        ureg[v]    = u[(size_t)m * DRES + col];
        holdreg[v] = (float)hr_hi[v * DRES + col] + (float)hr_lo[v * DRES + col];
        if (t + 1 < TSTEPS) __builtin_prefetch(&u[(size_t)(m + 1) * DRES + col], 0, 1);
      }
    }

    v8f c = {};
#pragma unroll
    for (int kb16 = 0; kb16 < 16; ++kb16) {
      const int ko = k0 + kb16 * 32;
      v16bf ah = load_fragA(hr_hi, lane & 15, DRES, ko, lane);
      v16bf al = load_fragA(hr_lo, lane & 15, DRES, ko, lane);
      v16bf bl = load_fragB(Wlo, nrow, DRES, ko, lane);
      c = wmma_bf16(ah, whi[kb16], c);   // hi*hi
      c = wmma_bf16(ah, bl, c);          // hi*lo
      c = wmma_bf16(al, whi[kb16], c);   // lo*hi
    }
#pragma unroll
    for (int v = 0; v < 8; ++v) partial[wave][v][lane] = c[v];
    __syncthreads();

    if (fin) {                                   // lanes 0..15 hold batches 0..7 (rows v)
#pragma unroll
      for (int v = 0; v < 8; ++v) {
        const int m = v * TSTEPS + t;            // row b*T + t
        float pre  = c[v] + partial[wave + 1][v][lane] + ureg[v];
        float hnew = om * holdreg[v] + lk * tanhf(pre);
        bf16_t hh = (bf16_t)hnew;
        bf16_t hl = (bf16_t)(hnew - (float)hh);
        hw_hi[v * DRES + col] = hh;
        hw_lo[v * DRES + col] = hl;
        Hhi[(size_t)m * DRES + col] = hh;
        Hlo[(size_t)m * DRES + col] = hl;
      }
    }

    // ---- inter-workgroup step barrier ----
    __threadfence();                             // release h stores (global_wb)
    __syncthreads();
    if (incluster) {
      __builtin_amdgcn_s_cluster_barrier();      // s_barrier_signal/-wait -3
      __threadfence();
      __syncthreads();
    } else {
      if (tid == 0) {
        unsigned prev = __hip_atomic_fetch_add(&bar[0], 1u, __ATOMIC_ACQ_REL,
                                               __HIP_MEMORY_SCOPE_AGENT);
        if (prev == 15u) {
          __hip_atomic_store(&bar[0], 0u, __ATOMIC_RELAXED, __HIP_MEMORY_SCOPE_AGENT);
          __hip_atomic_fetch_add(&bar[1], 1u, __ATOMIC_RELEASE, __HIP_MEMORY_SCOPE_AGENT);
        }
        const unsigned target = (unsigned)t + 1u;
        while (__hip_atomic_load(&bar[1], __ATOMIC_ACQUIRE, __HIP_MEMORY_SCOPE_AGENT) < target) {
          __builtin_amdgcn_s_sleep(2);
        }
      }
      __syncthreads();
      __threadfence();                           // acquire: invalidate stale lines (global_inv)
    }
  }
}

// ---------------- host launch ----------------
extern "C" void kernel_launch(void* const* d_in, const int* in_sizes, int n_in,
                              void* d_out, int out_size, void* d_ws, size_t ws_size,
                              hipStream_t stream) {
  (void)in_sizes; (void)n_in; (void)out_size; (void)ws_size;
  const float* x     = (const float*)d_in[0];
  const float* W_in  = (const float*)d_in[1];
  const float* W_res = (const float*)d_in[2];
  const float* W_ro  = (const float*)d_in[3];
  const float* lrho  = (const float*)d_in[4];
  const float* leak  = (const float*)d_in[5];
  float* out = (float*)d_out;

  char* w = (char*)d_ws;
  size_t off = 0;
  float*    scalars = (float*)(w + off);    off += 256;
  unsigned* bar     = (unsigned*)(w + off); off += 256;
  bf16_t* h_hi    = (bf16_t*)(w + off); off += (size_t)2 * 16 * DRES * 2;
  bf16_t* h_lo    = (bf16_t*)(w + off); off += (size_t)2 * 16 * DRES * 2;
  bf16_t* Win_hi  = (bf16_t*)(w + off); off += (size_t)DRES * DMODEL * 2;
  bf16_t* Win_lo  = (bf16_t*)(w + off); off += (size_t)DRES * DMODEL * 2;
  bf16_t* Wres_hi = (bf16_t*)(w + off); off += (size_t)DRES * DRES * 2;
  bf16_t* Wres_lo = (bf16_t*)(w + off); off += (size_t)DRES * DRES * 2;
  bf16_t* Wro_hi  = (bf16_t*)(w + off); off += (size_t)DMODEL * DRES * 2;
  bf16_t* Wro_lo  = (bf16_t*)(w + off); off += (size_t)DMODEL * DRES * 2;
  bf16_t* x_hi    = (bf16_t*)(w + off); off += (size_t)MROWS * DMODEL * 2;
  bf16_t* x_lo    = (bf16_t*)(w + off); off += (size_t)MROWS * DMODEL * 2;
  bf16_t* H_hi    = (bf16_t*)(w + off); off += (size_t)MROWS * DRES * 2;
  bf16_t* H_lo    = (bf16_t*)(w + off); off += (size_t)MROWS * DRES * 2;
  float*  u       = (float*)(w + off);  off += (size_t)MROWS * DRES * 4;

  // 1) spectral radius + leak scalars
  spectral_kernel<<<1, 1024, 0, stream>>>(W_res, lrho, leak, scalars);

  // 2) precision-split conversions (W_res scaled by device-computed scale)
  int n;
  n = DRES * DMODEL;  cvt_split_kernel<<<(n + 255) / 256, 256, 0, stream>>>(W_in, Win_hi, Win_lo, n, nullptr);
  n = DRES * DRES;    cvt_split_kernel<<<(n + 255) / 256, 256, 0, stream>>>(W_res, Wres_hi, Wres_lo, n, scalars);
  n = DMODEL * DRES;  cvt_split_kernel<<<(n + 255) / 256, 256, 0, stream>>>(W_ro, Wro_hi, Wro_lo, n, nullptr);
  n = MROWS * DMODEL; cvt_split_kernel<<<(n + 255) / 256, 256, 0, stream>>>(x, x_hi, x_lo, n, nullptr);

  // 3) init h double-buffer (zeros incl. pad rows) + barrier
  scan_init_kernel<<<128, 256, 0, stream>>>(h_hi, h_lo, bar);

  // 4) u = x @ W_in^T  (M=32768, N=1024, K=512)
  {
    int waves = (MROWS / 16) * (DRES / 16);
    gemm_bf16split_kernel<<<waves / 8, 256, 0, stream>>>(x_hi, x_lo, Win_hi, Win_lo, u,
                                                         MROWS, DRES, DMODEL);
  }

  // 5) serial reservoir scan (16 persistent WGs)
  scan_kernel<<<16, 256, 0, stream>>>(Wres_hi, Wres_lo, u, h_hi, h_lo, H_hi, H_lo, scalars, bar);

  // 6) out = H @ W_readout^T  (M=32768, N=512, K=1024)
  {
    int waves = (MROWS / 16) * (DMODEL / 16);
    gemm_bf16split_kernel<<<waves / 8, 256, 0, stream>>>(H_hi, H_lo, Wro_hi, Wro_lo, out,
                                                         MROWS, DMODEL, DRES);
  }
}
